// QuantizedMoEExperts_11871289607000
// MI455X (gfx1250) — matrix operand order repaired and checked by reading, not compiled, not run on hardware
//
#include <hip/hip_runtime.h>
#include <hip/hip_bf16.h>
#include <stdint.h>

// Problem constants (match reference)
constexpr int E_  = 8;
constexpr int H_  = 2048;
constexpr int I_  = 4096;
constexpr int T_  = 1024;
constexpr int K_  = 2;
constexpr int GS_ = 128;
constexpr int MAXL = T_ * K_;      // max list entries per expert (token may route twice)

// Tiling
constexpr int MT = 64;             // token-rows per block tile
constexpr int NT = 64;             // output cols per block tile
constexpr int KT = 32;             // K slice per WMMA step

typedef __attribute__((ext_vector_type(16))) _Float16 v16h;
typedef __attribute__((ext_vector_type(8)))  _Float16 v8h;
typedef __attribute__((ext_vector_type(8)))  float    v8f;

__device__ __forceinline__ float gelu_tanh(float v) {
    float v3 = v * v * v;
    return 0.5f * v * (1.0f + tanhf(0.7978845608028654f * (v + 0.044715f * v3)));
}

__device__ __forceinline__ v16h frag_from_lds_A(const _Float16* rowp, int kd) {
    // A element j: K = kd + (j<8 ? j : j+8); contiguous 8-half (16B) chunks
    v8h lo = *(const v8h*)(rowp + kd);
    v8h hi = *(const v8h*)(rowp + kd + 16);
    v16h a;
#pragma unroll
    for (int j = 0; j < 8; ++j) { a[j] = lo[j]; a[j + 8] = hi[j]; }
    return a;
}

__device__ __forceinline__ v16h frag_from_lds_B(const _Float16* rowp, int ko) {
    // B element j: K = ko/2 + (j<8 ? j : j+8); contiguous 8-half (16B) chunks
    v8h lo = *(const v8h*)(rowp + (ko >> 1));
    v8h hi = *(const v8h*)(rowp + (ko >> 1) + 16);
    v16h b;
#pragma unroll
    for (int j = 0; j < 8; ++j) { b[j] = lo[j]; b[j + 8] = hi[j]; }
    return b;
}

__global__ __launch_bounds__(256) void zero_kernel(float* __restrict__ out, int* __restrict__ cnt) {
    int idx = blockIdx.x * 256 + threadIdx.x;
    if (idx < T_ * H_) out[idx] = 0.0f;
    if (blockIdx.x == 0 && threadIdx.x < E_) cnt[threadIdx.x] = 0;
}

__global__ __launch_bounds__(256) void route_kernel(const int* __restrict__ ridx,
                                                    int* __restrict__ cnt,
                                                    int* __restrict__ list) {
    for (int i = threadIdx.x; i < T_ * K_; i += 256) {
        int e = ridx[i];
        int pos = atomicAdd(&cnt[e], 1);
        list[e * MAXL + pos] = i;          // slot = t*K + k
    }
}

// gu = x @ gu_w^T ; h = gelu(gate) * up  -> h_buf[slot][0..I)
__global__ __launch_bounds__(256)
void gu_kernel(const float* __restrict__ x,
               const uint8_t* __restrict__ gu_packed,
               const float* __restrict__ gu_scales,
               const float* __restrict__ gu_zeros,
               const int* __restrict__ cnt,
               const int* __restrict__ list,
               _Float16* __restrict__ h_buf) {
    const int e    = blockIdx.z;
    const int cntE = cnt[e];
    const int mBase = blockIdx.y * MT;
    if (mBase >= cntE) return;
    const int nBase = blockIdx.x * NT;     // within [0, I)

    __shared__ __align__(16) _Float16 As[MT][KT];
    __shared__ __align__(16) _Float16 Bg[NT][KT];
    __shared__ __align__(16) _Float16 Bu[NT][KT];

    const int tid = threadIdx.x;
    // ---- A loader role: 64 rows x 32 halves; 8 halves per thread
    const int am = tid >> 2;               // 0..63
    const int ak = (tid & 3) * 8;          // 0,8,16,24
    int arow = mBase + am; if (arow >= cntE) arow = cntE - 1;   // clamp (select, no branch)
    const int aslot = list[e * MAXL + arow];
    const float* xrow = x + (size_t)(aslot >> 1) * H_;          // K_ == 2
    // ---- B loader role: 64 rows x 16 packed bytes; 4 bytes per thread (x2 for gate/up)
    const int bn = tid >> 2;               // 0..63
    const int bk = (tid & 3) * 4;          // packed-byte offset 0,4,8,12
    const size_t gRow = (size_t)e * (2 * I_) + (size_t)(nBase + bn);
    const size_t uRow = gRow + I_;
    const uint8_t* gPack = gu_packed + gRow * (H_ / 2);
    const uint8_t* uPack = gu_packed + uRow * (H_ / 2);
    const float* gScale = gu_scales + gRow * (H_ / GS_);
    const float* gZero  = gu_zeros  + gRow * (H_ / GS_);
    const float* uScale = gu_scales + uRow * (H_ / GS_);
    const float* uZero  = gu_zeros  + uRow * (H_ / GS_);

    // ---- wave / lane mapping: 8 waves; each wave = one 16-row sub-tile x two 16-col sub-tiles
    const int lane = tid & 31;
    const int wv   = tid >> 5;
    const int msub = (wv >> 1) * 16;       // 0,16,32,48
    const int n0   = (wv & 1) * 32;        // 0 or 32; second sub-tile at n0+16
    const int lrow = lane & 15;
    const int lhi  = lane >> 4;            // 0/1
    const int kd   = lhi * 8;
    const int ko   = lhi * 16;

    v8f cg0 = {}, cg1 = {}, cu0 = {}, cu1 = {};

    for (int k0 = 0; k0 < H_; k0 += KT) {
        // stage A (f32 -> f16)
        float4 x0 = *(const float4*)(xrow + k0 + ak);
        float4 x1 = *(const float4*)(xrow + k0 + ak + 4);
        As[am][ak + 0] = (_Float16)x0.x; As[am][ak + 1] = (_Float16)x0.y;
        As[am][ak + 2] = (_Float16)x0.z; As[am][ak + 3] = (_Float16)x0.w;
        As[am][ak + 4] = (_Float16)x1.x; As[am][ak + 5] = (_Float16)x1.y;
        As[am][ak + 6] = (_Float16)x1.z; As[am][ak + 7] = (_Float16)x1.w;
        // stage B: dequant int4 -> f16 (high nibble = even col, low nibble = odd col)
        const int g = k0 >> 7;             // group of 128
        uint32_t gw = *(const uint32_t*)(gPack + (k0 >> 1) + bk);
        uint32_t uw = *(const uint32_t*)(uPack + (k0 >> 1) + bk);
        if (k0 + KT < H_) {                // prefetch next weight slice (global_prefetch_b8)
            __builtin_prefetch(gPack + ((k0 + KT) >> 1) + bk, 0, 1);
            __builtin_prefetch(uPack + ((k0 + KT) >> 1) + bk, 0, 1);
        }
        float gs = gScale[g], gz = gZero[g];
        float us = uScale[g], uz = uZero[g];
#pragma unroll
        for (int b = 0; b < 4; ++b) {
            uint32_t gb = (gw >> (8 * b)) & 0xFF;
            uint32_t ub = (uw >> (8 * b)) & 0xFF;
            int kk = 2 * (bk + b);
            Bg[bn][kk]     = (_Float16)((float)(gb >> 4) * gs + gz);
            Bg[bn][kk + 1] = (_Float16)((float)(gb & 15) * gs + gz);
            Bu[bn][kk]     = (_Float16)((float)(ub >> 4) * us + uz);
            Bu[bn][kk + 1] = (_Float16)((float)(ub & 15) * us + uz);
        }
        __syncthreads();

        v16h a   = frag_from_lds_A(&As[msub + lrow][0], kd);
        v16h bgA = frag_from_lds_B(&Bg[n0 + lrow][0], ko);
        v16h bgB = frag_from_lds_B(&Bg[n0 + 16 + lrow][0], ko);
        v16h buA = frag_from_lds_B(&Bu[n0 + lrow][0], ko);
        v16h buB = frag_from_lds_B(&Bu[n0 + 16 + lrow][0], ko);
        cg0 = __builtin_amdgcn_wmma_f32_16x16x32_f16(false, a, false, bgA, (short)0, cg0, false, false);
        cg1 = __builtin_amdgcn_wmma_f32_16x16x32_f16(false, a, false, bgB, (short)0, cg1, false, false);
        cu0 = __builtin_amdgcn_wmma_f32_16x16x32_f16(false, a, false, buA, (short)0, cu0, false, false);
        cu1 = __builtin_amdgcn_wmma_f32_16x16x32_f16(false, a, false, buB, (short)0, cu1, false, false);
        __syncthreads();
    }

    // epilogue: h = gelu(gate) * up   (C/D layout: VGPR r -> M = r + 8*lhi, col = lane%16)
    const int nG0 = nBase + n0 + lrow;
    const int nG1 = nG0 + 16;
#pragma unroll
    for (int r = 0; r < 8; ++r) {
        int row = mBase + msub + r + 8 * lhi;
        if (row < cntE) {
            int slot = list[e * MAXL + row];
            _Float16* hp = h_buf + (size_t)slot * I_;
            hp[nG0] = (_Float16)(gelu_tanh(cg0[r]) * cu0[r]);
            hp[nG1] = (_Float16)(gelu_tanh(cg1[r]) * cu1[r]);
        }
    }
}

// out[t] += coef * (h @ dn_w^T)
__global__ __launch_bounds__(256)
void dn_kernel(const _Float16* __restrict__ h_buf,
               const uint8_t* __restrict__ dn_packed,
               const float* __restrict__ dn_scales,
               const float* __restrict__ dn_zeros,
               const float* __restrict__ rweights,
               const int* __restrict__ cnt,
               const int* __restrict__ list,
               float* __restrict__ out) {
    const int e    = blockIdx.z;
    const int cntE = cnt[e];
    const int mBase = blockIdx.y * MT;
    if (mBase >= cntE) return;
    const int nBase = blockIdx.x * NT;     // within [0, H)

    __shared__ __align__(16) _Float16 As[MT][KT];
    __shared__ __align__(16) _Float16 Bw[NT][KT];

    const int tid = threadIdx.x;
    const int am = tid >> 2;
    const int ak = (tid & 3) * 8;
    int arow = mBase + am; if (arow >= cntE) arow = cntE - 1;
    const int aslot = list[e * MAXL + arow];
    const _Float16* hrow = h_buf + (size_t)aslot * I_;

    const int bn = tid >> 2;
    const int bk = (tid & 3) * 4;
    const size_t wRow = (size_t)e * H_ + (size_t)(nBase + bn);
    const uint8_t* wPack = dn_packed + wRow * (I_ / 2);
    const float* wScale = dn_scales + wRow * (I_ / GS_);
    const float* wZero  = dn_zeros  + wRow * (I_ / GS_);

    const int lane = tid & 31;
    const int wv   = tid >> 5;
    const int msub = (wv >> 1) * 16;
    const int n0   = (wv & 1) * 32;
    const int lrow = lane & 15;
    const int lhi  = lane >> 4;
    const int kd   = lhi * 8;
    const int ko   = lhi * 16;

    v8f c0 = {}, c1 = {};

    for (int k0 = 0; k0 < I_; k0 += KT) {
        *(v8h*)&As[am][ak] = *(const v8h*)(hrow + k0 + ak);   // already f16
        const int g = k0 >> 7;
        uint32_t ww = *(const uint32_t*)(wPack + (k0 >> 1) + bk);
        if (k0 + KT < I_) __builtin_prefetch(wPack + ((k0 + KT) >> 1) + bk, 0, 1);
        float ws = wScale[g], wz = wZero[g];
#pragma unroll
        for (int b = 0; b < 4; ++b) {
            uint32_t wb = (ww >> (8 * b)) & 0xFF;
            int kk = 2 * (bk + b);
            Bw[bn][kk]     = (_Float16)((float)(wb >> 4) * ws + wz);
            Bw[bn][kk + 1] = (_Float16)((float)(wb & 15) * ws + wz);
        }
        __syncthreads();

        v16h a  = frag_from_lds_A(&As[msub + lrow][0], kd);
        v16h b0 = frag_from_lds_B(&Bw[n0 + lrow][0], ko);
        v16h b1 = frag_from_lds_B(&Bw[n0 + 16 + lrow][0], ko);
        c0 = __builtin_amdgcn_wmma_f32_16x16x32_f16(false, a, false, b0, (short)0, c0, false, false);
        c1 = __builtin_amdgcn_wmma_f32_16x16x32_f16(false, a, false, b1, (short)0, c1, false, false);
        __syncthreads();
    }

    const int nG0 = nBase + n0 + lrow;
    const int nG1 = nG0 + 16;
#pragma unroll
    for (int r = 0; r < 8; ++r) {
        int row = mBase + msub + r + 8 * lhi;
        if (row < cntE) {
            int slot = list[e * MAXL + row];
            float coef = rweights[slot];
            float* op = out + (size_t)(slot >> 1) * H_;
            atomicAdd(op + nG0, coef * c0[r]);   // exactly 2 adds per element: order-invariant
            atomicAdd(op + nG1, coef * c1[r]);
        }
    }
}

extern "C" void kernel_launch(void* const* d_in, const int* in_sizes, int n_in,
                              void* d_out, int out_size, void* d_ws, size_t ws_size,
                              hipStream_t stream) {
    (void)in_sizes; (void)n_in; (void)out_size; (void)ws_size;
    const float*   x       = (const float*)d_in[0];
    const uint8_t* gu_pack = (const uint8_t*)d_in[1];
    const float*   gu_s    = (const float*)d_in[2];
    const float*   gu_z    = (const float*)d_in[3];
    const uint8_t* dn_pack = (const uint8_t*)d_in[4];
    const float*   dn_s    = (const float*)d_in[5];
    const float*   dn_z    = (const float*)d_in[6];
    const int*     ridx    = (const int*)d_in[7];
    const float*   rw      = (const float*)d_in[8];
    float* out = (float*)d_out;

    char* ws = (char*)d_ws;
    int* cnt  = (int*)ws;                              // 8 ints
    int* list = (int*)(ws + 256);                      // E*MAXL ints = 64 KB
    _Float16* h_buf = (_Float16*)(ws + (128 * 1024));  // T*K*I f16 = 16 MiB

    zero_kernel<<<(T_ * H_ + 255) / 256, 256, 0, stream>>>(out, cnt);
    route_kernel<<<1, 256, 0, stream>>>(ridx, cnt, list);

    dim3 gGrid(I_ / NT, (MAXL + MT - 1) / MT, E_);     // 64 x 32 x 8 (early-exit on cnt)
    gu_kernel<<<gGrid, 256, 0, stream>>>(x, gu_pack, gu_s, gu_z, cnt, list, h_buf);

    dim3 dGrid(H_ / NT, (MAXL + MT - 1) / MT, E_);     // 32 x 32 x 8
    dn_kernel<<<dGrid, 256, 0, stream>>>(h_buf, dn_pack, dn_s, dn_z, rw, cnt, list, out);
}
